// GraphQualityMetric_35811437314538
// MI455X (gfx1250) — matrix operand order repaired and checked by reading, not compiled, not run on hardware
//
#include <hip/hip_runtime.h>
#include <hip/hip_bf16.h>

#define GN 2048

typedef __attribute__((ext_vector_type(16))) __bf16 v16bf;
typedef __attribute__((ext_vector_type(8)))  __bf16 v8bf;
typedef __attribute__((ext_vector_type(8)))  float  v8f;

__device__ __forceinline__ v16bf cat16(v8bf a, v8bf b) {
    v16bf r;
#pragma unroll
    for (int i = 0; i < 8; i++) { r[i] = a[i]; r[i + 8] = b[i]; }
    return r;
}

// ---------------------------------------------------------------------------
// Prep: fp32 A -> bf16 hi/lo (row-major) + bf16 hi/lo transposed (via LDS tile)
// block (32,8), each block handles one 32x32 tile.
// ---------------------------------------------------------------------------
__global__ void prep_kernel(const float* __restrict__ A,
                            __bf16* __restrict__ Ahi,  __bf16* __restrict__ Alo,
                            __bf16* __restrict__ AThi, __bf16* __restrict__ ATlo)
{
    __shared__ float tile[32][33];
    const int bx = blockIdx.x, by = blockIdx.y;
    const int tx = threadIdx.x, ty = threadIdx.y;

#pragma unroll
    for (int p = 0; p < 4; p++) {
        const int r   = ty + p * 8;
        const int row = by * 32 + r;
        const int col = bx * 32 + tx;
        const float a = A[(size_t)row * GN + col];
        const __bf16 h = (__bf16)a;
        const __bf16 l = (__bf16)(a - (float)h);
        Ahi[(size_t)row * GN + col] = h;
        Alo[(size_t)row * GN + col] = l;
        tile[r][tx] = a;
    }
    __syncthreads();
#pragma unroll
    for (int p = 0; p < 4; p++) {
        const int r    = ty + p * 8;
        const int orow = bx * 32 + r;   // = original column
        const int ocol = by * 32 + tx;  // = original row
        const float a  = tile[tx][r];   // A[ocol][orow]
        const __bf16 h = (__bf16)a;
        const __bf16 l = (__bf16)(a - (float)h);
        AThi[(size_t)orow * GN + ocol] = h;
        ATlo[(size_t)orow * GN + ocol] = l;
    }
}

// ---------------------------------------------------------------------------
// Degree: one wave32 per row.
// ---------------------------------------------------------------------------
__global__ void deg_kernel(const float* __restrict__ A, float* __restrict__ deg)
{
    const int wave = threadIdx.x >> 5;
    const int lane = threadIdx.x & 31;
    const int row  = blockIdx.x * 8 + wave;
    const float* rp = A + (size_t)row * GN;
    float s = 0.0f;
    for (int c = lane; c < GN; c += 32) s += rp[c];
#pragma unroll
    for (int m = 1; m < 32; m <<= 1) s += __shfl_xor(s, m, 32);
    if (lane == 0) deg[row] = s;
}

__global__ void zero_kernel(float* __restrict__ p, int n)
{
    const int i = blockIdx.x * blockDim.x + threadIdx.x;
    if (i < n) p[i] = 0.0f;
}

// ---------------------------------------------------------------------------
// tri[i] = sum_j (A@A)[i,j] * A[j,i] via bf16 split-precision WMMA.
// One wave (block of 32) computes a 16x64 strip of A@A (4 accumulators),
// K swept 0..2048 in steps of 32, 3 WMMAs per K-step per tile
// (hi*hi + hi*lo + lo*hi ~= fp32 product).
// grid = (GN/16, GN/64), block = 32.
// ---------------------------------------------------------------------------
__global__ void __launch_bounds__(32)
tri_kernel(const float*  __restrict__ A,
           const __bf16* __restrict__ Ahi,  const __bf16* __restrict__ Alo,
           const __bf16* __restrict__ AThi, const __bf16* __restrict__ ATlo,
           float* __restrict__ tri)
{
    const int lane = threadIdx.x;
    const int half = lane >> 4;       // 0: lanes 0-15, 1: lanes 16-31
    const int l16  = lane & 15;
    const int m0   = blockIdx.x * 16; // output row tile
    const int nb   = blockIdx.y * 64; // output column strip (4 tiles of 16)

    v8f acc[4] = {};

    // A-operand row for this lane (16 contiguous K values / lane, matching B)
    const int mrow = m0 + l16;
    const __bf16* arow_hi = Ahi + (size_t)mrow * GN;
    const __bf16* arow_lo = Alo + (size_t)mrow * GN;

    for (int k0 = 0; k0 < GN; k0 += 32) {
        const v16bf a_hi = cat16(*(const v8bf*)(arow_hi + k0 + half * 16),
                                 *(const v8bf*)(arow_hi + k0 + half * 16 + 8));
        const v16bf a_lo = cat16(*(const v8bf*)(arow_lo + k0 + half * 16),
                                 *(const v8bf*)(arow_lo + k0 + half * 16 + 8));
#pragma unroll
        for (int t = 0; t < 4; t++) {
            // B-operand column for this lane: 16 contiguous K values from A^T
            const int ncol = nb + t * 16 + l16;
            const __bf16* bp_hi = AThi + (size_t)ncol * GN + k0 + half * 16;
            const __bf16* bp_lo = ATlo + (size_t)ncol * GN + k0 + half * 16;
            const v16bf b_hi = cat16(*(const v8bf*)(bp_hi), *(const v8bf*)(bp_hi + 8));
            const v16bf b_lo = cat16(*(const v8bf*)(bp_lo), *(const v8bf*)(bp_lo + 8));
            acc[t] = __builtin_amdgcn_wmma_f32_16x16x32_bf16(
                false, a_hi, false, b_hi, (short)0, acc[t], false, false);
            acc[t] = __builtin_amdgcn_wmma_f32_16x16x32_bf16(
                false, a_hi, false, b_lo, (short)0, acc[t], false, false);
            acc[t] = __builtin_amdgcn_wmma_f32_16x16x32_bf16(
                false, a_lo, false, b_hi, (short)0, acc[t], false, false);
        }
    }

    // Epilogue: elementwise * A^T (exact fp32 from original graph), reduce over
    // N within each 16-lane half, atomically accumulate per-row triangles.
#pragma unroll
    for (int t = 0; t < 4; t++) {
        const int nrow = nb + t * 16 + l16;
        // A[nrow, m0+8*half .. +7] : 8 contiguous fp32, 32B aligned
        const float* wrow = A + (size_t)nrow * GN + m0 + half * 8;
        const float4 w0 = *(const float4*)(wrow);
        const float4 w1 = *(const float4*)(wrow + 4);
        float w[8] = { w0.x, w0.y, w0.z, w0.w, w1.x, w1.y, w1.z, w1.w };
#pragma unroll
        for (int r = 0; r < 8; r++) {
            float p = acc[t][r] * w[r];        // C[m0+r+8*half, nrow] * A[nrow, m]
            p += __shfl_xor(p, 1, 32);
            p += __shfl_xor(p, 2, 32);
            p += __shfl_xor(p, 4, 32);
            p += __shfl_xor(p, 8, 32);         // sum over the 16-lane half (N)
            if (l16 == 0) atomicAdd(&tri[m0 + r + half * 8], p);
        }
    }
}

// ---------------------------------------------------------------------------
// Finalize: degree_sim, cluster_sim, overall. Single block.
// ---------------------------------------------------------------------------
__global__ void finalize_kernel(const float* __restrict__ deg1, const float* __restrict__ deg2,
                                const float* __restrict__ tri1, const float* __restrict__ tri2,
                                float* __restrict__ out)
{
    __shared__ float sd[256];
    __shared__ float sc[256];
    const int tid = threadIdx.x;
    float ad = 0.0f, ac = 0.0f;
    for (int i = tid; i < GN; i += 256) {
        const float d1 = deg1[i], d2 = deg2[i];
        ad += fabsf(d1 - d2);
        const float dd1 = (d1 < 2.0f) ? 1.0f : d1;
        const float dd2 = (d2 < 2.0f) ? 1.0f : d2;
        const float c1 = tri1[i] / (dd1 * (dd1 - 1.0f));
        const float c2 = tri2[i] / (dd2 * (dd2 - 1.0f));
        ac += fabsf(c1 - c2);
    }
    sd[tid] = ad; sc[tid] = ac;
    __syncthreads();
    for (int s = 128; s > 0; s >>= 1) {
        if (tid < s) { sd[tid] += sd[tid + s]; sc[tid] += sc[tid + s]; }
        __syncthreads();
    }
    if (tid == 0) {
        const float degree_sim  = expf(-sd[0] / (float)GN);
        const float cluster_sim = expf(-sc[0] / (float)GN);
        out[0] = degree_sim;
        out[1] = cluster_sim;
        out[2] = 0.5f * (degree_sim + cluster_sim);
    }
}

// ---------------------------------------------------------------------------
extern "C" void kernel_launch(void* const* d_in, const int* in_sizes, int n_in,
                              void* d_out, int out_size, void* d_ws, size_t ws_size,
                              hipStream_t stream)
{
    const float* A1 = (const float*)d_in[0];
    const float* A2 = (const float*)d_in[1];
    float* out = (float*)d_out;

    const size_t SZ   = (size_t)GN * (size_t)GN;   // elements per matrix
    const size_t BF   = SZ * 2;                    // bytes per bf16 matrix
    char* ws = (char*)d_ws;

    __bf16* Ahi1  = (__bf16*)(ws + 0 * BF);
    __bf16* Alo1  = (__bf16*)(ws + 1 * BF);
    __bf16* AThi1 = (__bf16*)(ws + 2 * BF);
    __bf16* ATlo1 = (__bf16*)(ws + 3 * BF);
    __bf16* Ahi2  = (__bf16*)(ws + 4 * BF);
    __bf16* Alo2  = (__bf16*)(ws + 5 * BF);
    __bf16* AThi2 = (__bf16*)(ws + 6 * BF);
    __bf16* ATlo2 = (__bf16*)(ws + 7 * BF);

    float* deg1 = (float*)(ws + 8 * BF);
    float* deg2 = deg1 + GN;
    float* tri1 = deg2 + GN;
    float* tri2 = tri1 + GN;

    // 1) bf16 hi/lo split + transpose copies
    dim3 pb(32, 8);
    dim3 pg(GN / 32, GN / 32);
    prep_kernel<<<pg, pb, 0, stream>>>(A1, Ahi1, Alo1, AThi1, ATlo1);
    prep_kernel<<<pg, pb, 0, stream>>>(A2, Ahi2, Alo2, AThi2, ATlo2);

    // 2) degree distributions (row sums, fp32)
    deg_kernel<<<GN / 8, 256, 0, stream>>>(A1, deg1);
    deg_kernel<<<GN / 8, 256, 0, stream>>>(A2, deg2);

    // 3) zero triangle accumulators (tri1 and tri2 are contiguous)
    zero_kernel<<<(2 * GN + 255) / 256, 256, 0, stream>>>(tri1, 2 * GN);

    // 4) WMMA triangle counts: tri_i = sum_j (A@A)_{ij} A_{ji}
    dim3 tg(GN / 16, GN / 64);
    tri_kernel<<<tg, 32, 0, stream>>>(A1, Ahi1, Alo1, AThi1, ATlo1, tri1);
    tri_kernel<<<tg, 32, 0, stream>>>(A2, Ahi2, Alo2, AThi2, ATlo2, tri2);

    // 5) similarities + overall
    finalize_kernel<<<1, 256, 0, stream>>>(deg1, deg2, tri1, tri2, out);
}
